// SpatialSoftmax_50173807952795
// MI455X (gfx1250) — compile-verified, matrix operand order checked by reading, not run on hardware
//
#include <hip/hip_runtime.h>
#include <hip/hip_bf16.h>

typedef __attribute__((ext_vector_type(16))) _Float16 v16h;
typedef __attribute__((ext_vector_type(8)))  float    v8f;

// ---------------------------------------------------------------------------
// Pass 1: s[b*N+n] = sum_c exp(z[b,n,c]) via WMMA ones-GEMV.
// One wave32 handles a tile of 16 flattened positions x 256 channels:
//   8 x v_wmma_f32_16x16x32_f16 accumulations with B == all-ones.
// Since B is all-ones, D[m][n] = sum_k A[m][k] regardless of K placement, so
// each lane just supplies 16 distinct channel values of its row:
//   lane <16  -> position = lane,    channels [chunk*32 +  0 .. +15]
//   lane >=16 -> position = lane-16, channels [chunk*32 + 16 .. +31]
// D layout (16x16 f32): VGPR r, lanes 0-15 hold M=r, lanes 16-31 hold M=8+r,
// every column identical -> lane 0 extracts M=0..7, lane 16 extracts M=8..15.
// ---------------------------------------------------------------------------
__global__ void __launch_bounds__(256)
channel_sum_wmma_kernel(const float* __restrict__ z,
                        float* __restrict__ s,
                        int C) {
    const int lane        = threadIdx.x & 31;
    const int waveInBlock = threadIdx.x >> 5;
    const int waveId      = blockIdx.x * (blockDim.x >> 5) + waveInBlock;

    const long pos0  = (long)waveId * 16;          // first of 16 positions
    const int  m     = lane & 15;                  // row of A this lane feeds
    const int  kHalf = (lane >> 4) * 16;           // 0 or 16 inside 32-chunk

    const float* __restrict__ row = z + (pos0 + m) * (long)C;

    v16h ones;
#pragma unroll
    for (int i = 0; i < 16; ++i) ones[i] = (_Float16)1.0f;

    v8f acc = {};

#pragma unroll
    for (int chunk = 0; chunk < 8; ++chunk) {
        const float4* __restrict__ p4 =
            reinterpret_cast<const float4*>(row + chunk * 32 + kHalf);
        float v[16];
        *reinterpret_cast<float4*>(&v[0])  = p4[0];
        *reinterpret_cast<float4*>(&v[4])  = p4[1];
        *reinterpret_cast<float4*>(&v[8])  = p4[2];
        *reinterpret_cast<float4*>(&v[12]) = p4[3];

        v16h aH;
#pragma unroll
        for (int i = 0; i < 16; ++i) aH[i] = (_Float16)__expf(v[i]);

        // D = A x ones + acc   (emits v_wmma_f32_16x16x32_f16)
        acc = __builtin_amdgcn_wmma_f32_16x16x32_f16(
            /*neg_a=*/false, aH, /*neg_b=*/false, ones,
            /*c_mod=*/(short)0, acc, /*reuse_a=*/false, /*reuse_b=*/false);
    }

    if (lane == 0) {
#pragma unroll
        for (int r = 0; r < 8; ++r) s[pos0 + r] = acc[r];
    } else if (lane == 16) {
#pragma unroll
        for (int r = 0; r < 8; ++r) s[pos0 + 8 + r] = acc[r];
    }
}

// ---------------------------------------------------------------------------
// Pass 2: denom[b,n] = zero-padded width-window sum of s along n.
// Tiny (B*N = 65536 elements); s sits in L2. width read on-device (no host
// sync needed, harness passes scalars as 1-element device arrays).
// ---------------------------------------------------------------------------
__global__ void __launch_bounds__(256)
window_sum_kernel(const float* __restrict__ s,
                  const int* __restrict__ width_p,
                  float* __restrict__ denom,
                  int N) {
    const int idx = blockIdx.x * blockDim.x + threadIdx.x;   // over B*N
    const int b = idx / N;
    const int n = idx - b * N;
    const int half = (*width_p) >> 1;

    const float* __restrict__ sb = s + (long)b * N;
    float acc = 0.0f;
    for (int j = -half; j <= half; ++j) {
        const int nn = n + j;
        if (nn >= 0 && nn < N) acc += sb[nn];
    }
    denom[idx] = acc;
}

// ---------------------------------------------------------------------------
// Pass 3: out = z * exp(z) / denom. One wave per position (32 lanes x 8 ch),
// float4 vector loads/stores, one reciprocal per thread.
// ---------------------------------------------------------------------------
__global__ void __launch_bounds__(256)
finalize_kernel(const float* __restrict__ z,
                const float* __restrict__ denom,
                float* __restrict__ out,
                int C) {
    const long tid      = (long)blockIdx.x * blockDim.x + threadIdx.x;
    const int  perPos   = C >> 3;                 // threads per position (32)
    const long pos      = tid / perPos;
    const int  cOff     = (int)(tid - pos * perPos) << 3;

    const float inv = __frcp_rn(denom[pos]);

    const float4* __restrict__ zp =
        reinterpret_cast<const float4*>(z + pos * (long)C + cOff);
    float4* __restrict__ op =
        reinterpret_cast<float4*>(out + pos * (long)C + cOff);

    float4 z0 = zp[0];
    float4 z1 = zp[1];

    float4 o0, o1;
    o0.x = z0.x * __expf(z0.x) * inv;
    o0.y = z0.y * __expf(z0.y) * inv;
    o0.z = z0.z * __expf(z0.z) * inv;
    o0.w = z0.w * __expf(z0.w) * inv;
    o1.x = z1.x * __expf(z1.x) * inv;
    o1.y = z1.y * __expf(z1.y) * inv;
    o1.z = z1.z * __expf(z1.z) * inv;
    o1.w = z1.w * __expf(z1.w) * inv;

    op[0] = o0;
    op[1] = o1;
}

extern "C" void kernel_launch(void* const* d_in, const int* in_sizes, int n_in,
                              void* d_out, int out_size, void* d_ws, size_t ws_size,
                              hipStream_t stream) {
    constexpr int B = 8, N = 8192, C = 256;

    const float* z       = (const float*)d_in[0];
    const int*   width_p = (const int*)d_in[1];
    float*       out     = (float*)d_out;

    float* s     = (float*)d_ws;                  // B*N floats  (256 KB)
    float* denom = s + (size_t)B * N;             // B*N floats  (256 KB)

    // Pass 1: 65536 positions / 16 per wave = 4096 waves; 8 waves per block.
    {
        const int waves  = (B * N) / 16;
        const int blocks = waves / 8;             // 512
        channel_sum_wmma_kernel<<<blocks, 256, 0, stream>>>(z, s, C);
    }

    // Pass 2: one thread per (b, n).
    {
        const int total  = B * N;
        const int blocks = total / 256;           // 256
        window_sum_kernel<<<blocks, 256, 0, stream>>>(s, width_p, denom, N);
    }

    // Pass 3: 8 floats per thread.
    {
        const long total  = (long)B * N * C / 8;  // 2,097,152
        const int  blocks = (int)(total / 256);   // 8192
        finalize_kernel<<<blocks, 256, 0, stream>>>(z, denom, out, C);
    }
}
